// MultiInputLSTMCell_39745627357368
// MI455X (gfx1250) — matrix operand order repaired and checked by reading, not compiled, not run on hardware
//
#include <hip/hip_runtime.h>
#include <hip/hip_bf16.h>

typedef __attribute__((ext_vector_type(16))) _Float16 v16h;
typedef __attribute__((ext_vector_type(8)))  float    v8f;

#define BATCH 65536
#define HID   256

// ---------------------------------------------------------------------------
// Weight pre-pack: f32 -> f16, swizzled into the WMMA B-fragment layout.
// Packed half index: ((((cu*8 + sec)*8 + kb)*32 + lane)*16 + e)
//   cu  (0..15): 16-column unit of the output
//   sec (0..7):  0:Wih(i) 1:Whh(i) 2:Wih(o) 3:Whh(o) 4:Wih(g) 5:Whh(g)
//                6:aWih   7:aWhh
//   kb  (0..7):  K-block of 32 (K = 256)
//   B layout (K x N = 32 x 16, f16): lane l holds col N = l%16,
//     K = kb*32 + (e/8)*16 + (l/16)*8 + (e%8),  e = 0..15.
// ---------------------------------------------------------------------------
__global__ __launch_bounds__(256) void pack_weights_kernel(
    const float* __restrict__ wih,  const float* __restrict__ whh,
    const float* __restrict__ awih, const float* __restrict__ awhh,
    _Float16* __restrict__ packed)
{
    const int idx  = blockIdx.x * 256 + threadIdx.x;   // 32*8*8*32*16 = 1048576
    const int e    = idx & 15;
    const int lane = (idx >> 4) & 31;
    const int kb   = (idx >> 9) & 7;
    const int sec  = (idx >> 12) & 7;
    const int cu   = idx >> 15;

    const int K = kb * 32 + (e >> 3) * 16 + (lane >> 4) * 8 + (e & 7);
    const int n = cu * 16 + (lane & 15);

    float w;
    switch (sec) {
        case 0: w = wih [K * 768 + n      ]; break;  // i-gate
        case 1: w = whh [K * 768 + n      ]; break;
        case 2: w = wih [K * 768 + n + 256]; break;  // o-gate
        case 3: w = whh [K * 768 + n + 256]; break;
        case 4: w = wih [K * 768 + n + 512]; break;  // g-gate
        case 5: w = whh [K * 768 + n + 512]; break;
        case 6: w = awih[K * 256 + n      ]; break;  // alpha
        default:w = awhh[K * 256 + n      ]; break;
    }
    packed[idx] = (_Float16)w;
}

// ---------------------------------------------------------------------------
// CDNA5 async memory->LDS staging (ASYNCcnt path), per-lane b128 transfers.
// LDS destination operand is a wave-relative LDS byte address in a VGPR.
// ---------------------------------------------------------------------------
static __device__ inline void async_stage_64k(const _Float16* __restrict__ g,
                                              unsigned ldsBase, int tid) {
    #pragma unroll
    for (int i = 0; i < 16; ++i) {
        const void* gp = (const void*)((const char*)g + (size_t)(i * 256 + tid) * 16);
        unsigned lds = ldsBase + (unsigned)(i * 256 + tid) * 16u;
        asm volatile("global_load_async_to_lds_b128 %0, %1, off"
                     :: "v"(lds), "v"(gp) : "memory");
    }
}
#define WAIT_ASYNC() asm volatile("s_wait_asynccnt 0x0" ::: "memory")

// ---------------------------------------------------------------------------
// Fused LSTM-cell kernel
// ---------------------------------------------------------------------------
static __device__ inline v8f wmma_f16(v16h a, v16h b, v8f c) {
    return __builtin_amdgcn_wmma_f32_16x16x32_f16(
        /*neg_a=*/false, a, /*neg_b=*/false, b,
        /*c_mod=*/(short)0, c, /*reuse_a=*/false, /*reuse_b=*/false);
}

// A-fragment (16x32 f16): lane l holds row M=l%16; with kg=l/16:
//   elements 0..7  = K = kb*32 + kg*8      + e
//   elements 8..15 = K = kb*32 + 16 + kg*8 + (e-8)
static __device__ inline v16h load_afrag(const float* __restrict__ p) {
    float4 r0 = *(const float4*)(p);
    float4 r1 = *(const float4*)(p + 4);
    float4 r2 = *(const float4*)(p + 16);
    float4 r3 = *(const float4*)(p + 20);
    v16h a;
    a[0]=(_Float16)r0.x; a[1]=(_Float16)r0.y; a[2]=(_Float16)r0.z; a[3]=(_Float16)r0.w;
    a[4]=(_Float16)r1.x; a[5]=(_Float16)r1.y; a[6]=(_Float16)r1.z; a[7]=(_Float16)r1.w;
    a[8]=(_Float16)r2.x; a[9]=(_Float16)r2.y; a[10]=(_Float16)r2.z; a[11]=(_Float16)r2.w;
    a[12]=(_Float16)r3.x; a[13]=(_Float16)r3.y; a[14]=(_Float16)r3.z; a[15]=(_Float16)r3.w;
    return a;
}

static __device__ inline float sigmoid_fast(float x) {
    x = fminf(fmaxf(x, -30.f), 30.f);
    return 1.f / (1.f + __expf(-x));
}
static __device__ inline float tanh_fast(float x) {
    x = fminf(fmaxf(x, -10.f), 10.f);
    float t = __expf(2.f * x);
    return (t - 1.f) / (t + 1.f);
}

__global__ __launch_bounds__(256) void lstm_fused_kernel(
    const float* __restrict__ input_, const float* __restrict__ c_input,
    const float* __restrict__ h0,
    const float* __restrict__ bias,  const float* __restrict__ abias,
    const _Float16* __restrict__ packedW,
    float* __restrict__ out_h, float* __restrict__ out_c)
{
    extern __shared__ unsigned char smem[];   // 2 x 64KB double-buffered B slab

    const int tid  = threadIdx.x;
    const int wave = tid >> 5;
    const int lane = tid & 31;
    const int kg   = lane >> 4;        // 0 or 1: which K-half this lane holds
    const int nl   = lane & 15;        // column-within-tile

    const int rowBase = blockIdx.x * 128 + wave * 16;
    const int myRow   = rowBase + nl;  // A-fragment row for this lane

    // Kick off staging of the first B slab while we load A-fragments.
    async_stage_64k(packedW, 0u, tid);

    // ---- Load all A-fragments once (input_, h_0, c_in), f32 -> f16 --------
    v16h aIn[8], aH[8], aC[8];
    #pragma unroll
    for (int kb = 0; kb < 8; ++kb) {
        const int k0 = kb * 32 + kg * 8;
        aIn[kb] = load_afrag(input_  + (size_t)myRow * HID + k0);
        aH [kb] = load_afrag(h0      + (size_t)myRow * HID + k0);
        aC [kb] = load_afrag(c_input + (size_t)myRow * HID + k0);
    }

    const int rv0 = rowBase + kg * 8;  // first C/D row held by this lane

    WAIT_ASYNC();
    __syncthreads();

    for (int cu = 0; cu < 16; ++cu) {
        const unsigned cbuf = (unsigned)(cu & 1) * 65536u;

        // ---- Prefetch next B slab asynchronously into the other buffer ----
        if (cu < 15)
            async_stage_64k(packedW + (size_t)(cu + 1) * 32768,
                            (unsigned)((cu + 1) & 1) * 65536u, tid);

        // ---- 64 WMMAs: i / o / g / alpha tiles over K=512 -----------------
        v8f acc0 = {}, acc1 = {}, acc2 = {}, acc3 = {};
        const unsigned char* sb = smem + cbuf + lane * 32;
        #pragma unroll
        for (int kb = 0; kb < 8; ++kb) {
            const int kbo = kb * 1024;
            acc0 = wmma_f16(aIn[kb], *(const v16h*)(sb + 0*8192 + kbo), acc0);
            acc0 = wmma_f16(aH [kb], *(const v16h*)(sb + 1*8192 + kbo), acc0);
            acc1 = wmma_f16(aIn[kb], *(const v16h*)(sb + 2*8192 + kbo), acc1);
            acc1 = wmma_f16(aH [kb], *(const v16h*)(sb + 3*8192 + kbo), acc1);
            acc2 = wmma_f16(aIn[kb], *(const v16h*)(sb + 4*8192 + kbo), acc2);
            acc2 = wmma_f16(aH [kb], *(const v16h*)(sb + 5*8192 + kbo), acc2);
            acc3 = wmma_f16(aIn[kb], *(const v16h*)(sb + 6*8192 + kbo), acc3);
            acc3 = wmma_f16(aC [kb], *(const v16h*)(sb + 7*8192 + kbo), acc3);
        }

        // ---- Fused elementwise tail; write h_1 / c_1 directly -------------
        const int col = cu * 16 + nl;
        const float bi = bias[col], bo = bias[HID + col], bg = bias[2*HID + col];
        const float ba = abias[col];

        #pragma unroll
        for (int v = 0; v < 8; ++v) {
            const int row = rv0 + v;
            const size_t off = (size_t)row * HID + col;
            const float cin = c_input[off];          // full-precision c_in
            const float iv = sigmoid_fast(acc0[v] + bi);
            const float ov = sigmoid_fast(acc1[v] + bo);
            const float gv = tanh_fast  (acc2[v] + bg);
            const float av = sigmoid_fast(acc3[v] + ba);
            const float ei = __expf(iv);
            const float ea = __expf(av);
            const float inv = 1.f / (ei + ea);
            const float c1 = (ei * gv + ea * cin) * inv;
            const float h1 = ov * tanh_fast(c1);
            __builtin_nontemporal_store(h1, out_h + off);
            __builtin_nontemporal_store(c1, out_c + off);
        }

        // All of this wave's async loads landed; barrier makes every wave's
        // writes visible and protects the buffer being overwritten next.
        WAIT_ASYNC();
        __syncthreads();
    }
}

// ---------------------------------------------------------------------------
extern "C" void kernel_launch(void* const* d_in, const int* in_sizes, int n_in,
                              void* d_out, int out_size, void* d_ws, size_t ws_size,
                              hipStream_t stream) {
    const float* input_ = (const float*)d_in[0];
    const float* c_inp  = (const float*)d_in[1];
    const float* h0     = (const float*)d_in[2];
    // d_in[3] = c_0 : unused by the reference computation
    const float* wih    = (const float*)d_in[4];
    const float* whh    = (const float*)d_in[5];
    const float* awih   = (const float*)d_in[6];
    const float* awhh   = (const float*)d_in[7];
    const float* bias   = (const float*)d_in[8];
    const float* abias  = (const float*)d_in[9];

    float* out_h = (float*)d_out;
    float* out_c = out_h + (size_t)BATCH * HID;
    _Float16* packed = (_Float16*)d_ws;   // needs 2 MB of workspace

    // Pack 1,048,576 f16 weights into WMMA B-fragment order (L2-resident).
    pack_weights_kernel<<<4096, 256, 0, stream>>>(wih, whh, awih, awhh, packed);

    // 512 workgroups x 256 threads, 128 KB dynamic LDS (double-buffered B).
    lstm_fused_kernel<<<BATCH / 128, 256, 131072, stream>>>(
        input_, c_inp, h0, bias, abias, packed, out_h, out_c);
}
